// DPDARTSArchMaster_49804440765132
// MI455X (gfx1250) — compile-verified, matrix operand order checked by reading, not run on hardware
//
#include <hip/hip_runtime.h>
#include <math.h>

// ---------------- problem constants (static per the reference) ----------------
#define H       2048
#define FOURH   8192
#define NOPS    8
#define NNODES  7
#define LAYERW  ((size_t)FOURH * H)       // 16,777,216 elems per layer per matrix

// lstm_layer tiling
#define WAVES   16                        // waves per block (512 threads)
#define SLICE   (2 * H / WAVES)           // 256: K-slice per wave of concat K=4096
#define CHUNKS  (SLICE / 32)              // 8 WMMA K-chunks per wave

typedef __attribute__((ext_vector_type(16))) __bf16 v16bf;
typedef __attribute__((ext_vector_type(8)))  __bf16 v8bf;
typedef __attribute__((ext_vector_type(2)))  __bf16 v2bf;
typedef __attribute__((ext_vector_type(8)))  float  v8f;

__device__ __forceinline__ float sigf(float x) { return 1.0f / (1.0f + __expf(-x)); }

// 8-element A-fragment half loads (bf16 direct = one global_load_b128; fp32 = cvt inline)
__device__ __forceinline__ v8bf ld8(const __bf16* p) { return *(const v8bf*)p; }
__device__ __forceinline__ v8bf ld8(const float* p) {
  v8bf r;
#pragma unroll
  for (int i = 0; i < 8; ++i) r[i] = (__bf16)p[i];
  return r;
}

// ---------------- fp32 -> bf16 weight conversion (once per launch) ----------------
__global__ void cvt_bf16(const float* __restrict__ in, __bf16* __restrict__ out, size_t n) {
  size_t stride = (size_t)gridDim.x * blockDim.x * 8;
  for (size_t i = ((size_t)blockIdx.x * blockDim.x + threadIdx.x) * 8; i < n; i += stride) {
    v8bf o;
#pragma unroll
    for (int k = 0; k < 8; ++k) o[k] = (__bf16)in[i + k];
    *(v8bf*)(out + i) = o;
  }
}

// ---------------- state init: h=c=0, x = start token embedding ----------------
__global__ void init_state(const float* __restrict__ w_emb, float* __restrict__ x,
                           float* __restrict__ hbuf, float* __restrict__ cbuf,
                           float* __restrict__ scal) {
  int tid = blockIdx.x * blockDim.x + threadIdx.x;
  int n = gridDim.x * blockDim.x;
  for (int j = tid; j < H; j += n) {
    x[j] = w_emb[NOPS * H + j];
    hbuf[j] = 0.f; hbuf[H + j] = 0.f; hbuf[2 * H + j] = 0.f; hbuf[3 * H + j] = 0.f;
    cbuf[j] = 0.f; cbuf[H + j] = 0.f;
  }
  if (tid == 0) { scal[0] = 0.f; scal[1] = 0.f; }
}

// ---------------- fused LSTM cell: gates matvec (WMMA) + cell update ----------------
// Block: 512 threads = 16 waves.  Grid: 128 blocks, block wg owns rows j = wg*16..wg*16+15
// of ALL FOUR gates.  Waves 0..7 cover W_ih@x, waves 8..15 cover W_hh@h (K-slices of 256).
// Inner loop per K-chunk: 8x global_load_b128 (weights) + 2x ds_load_b128 (bf16 vector)
// + 4x v_wmma_f32_16x16x32_bf16. Vector is converted to bf16 once at staging time.
template <typename WT>
__global__ __launch_bounds__(512)
void lstm_layer(const WT* __restrict__ Wih, const WT* __restrict__ Whh,
                const float* __restrict__ bih, const float* __restrict__ bhh,
                const float* __restrict__ x, const float* __restrict__ h_in,
                float* __restrict__ c, float* __restrict__ h_out) {
  __shared__ __bf16 vec_s[WAVES][SLICE];    // 8 KB (bf16 vector slices)
  __shared__ float red_s[WAVES][4][16];     // 4 KB
  __shared__ float gate_s[4][16];

  const int tid = threadIdx.x;
  const int wave = tid >> 5, lane = tid & 31;
  const int wg = blockIdx.x;
  const bool is_ih = wave < (WAVES / 2);
  const int kmb = (is_ih ? wave : wave - WAVES / 2) * SLICE;  // k base within 2048
  const WT* __restrict__ W = is_ih ? Wih : Whh;
  const float* __restrict__ vec = is_ih ? x : h_in;

  // stage this wave's input-vector slice into LDS, converting to bf16 now so the
  // inner loop carries no v_cvt and only two ds_load_b128 per chunk
  {
    const float2* v2 = (const float2*)(vec + kmb);
    v2bf* ls = (v2bf*)vec_s[wave];
    for (int i = lane; i < SLICE / 2; i += 32) {
      float2 f = v2[i];
      v2bf p; p[0] = (__bf16)f.x; p[1] = (__bf16)f.y;
      ls[i] = p;
    }
  }

  const int mrow = lane & 15;               // A-matrix row within the 16-row tile
  const int klo = (lane < 16) ? 0 : 8;      // 16-bit A layout: low/high K octets per half-wave
  v8f acc[4] = {};

  const WT* rowp[4];
#pragma unroll
  for (int g = 0; g < 4; ++g)
    rowp[g] = W + (size_t)(g * H + wg * 16 + mrow) * H + kmb + klo;

#pragma unroll 2
  for (int kc = 0; kc < CHUNKS; ++kc) {
    // B fragment: bf16 vector replicated into all 16 columns (batch-1 matvec)
    const __bf16* vs = &vec_s[wave][kc * 32 + klo];
    v8bf blo = *(const v8bf*)vs;            // ds_load_b128
    v8bf bhi = *(const v8bf*)(vs + 16);     // ds_load_b128
    v16bf b;
#pragma unroll
    for (int i = 0; i < 8; ++i) { b[i] = blo[i]; b[i + 8] = bhi[i]; }

#pragma unroll
    for (int g = 0; g < 4; ++g) {
      const WT* ap = rowp[g] + kc * 32;
      __builtin_prefetch(ap + 128, 0, 3);   // near-scope prefetch down the weight stream
      v8bf alo = ld8(ap);
      v8bf ahi = ld8(ap + 16);
      v16bf a;
#pragma unroll
      for (int i = 0; i < 8; ++i) { a[i] = alo[i]; a[i + 8] = ahi[i]; }
      acc[g] = __builtin_amdgcn_wmma_f32_16x16x32_bf16(
          false, a, false, b, (short)0, acc[g], false, false);
    }
  }

  // D columns are identical (B replicated). lane0 holds rows 0..7, lane16 rows 8..15.
  if (lane == 0) {
#pragma unroll
    for (int g = 0; g < 4; ++g)
#pragma unroll
      for (int r = 0; r < 8; ++r) red_s[wave][g][r] = acc[g][r];
  } else if (lane == 16) {
#pragma unroll
    for (int g = 0; g < 4; ++g)
#pragma unroll
      for (int r = 0; r < 8; ++r) red_s[wave][g][8 + r] = acc[g][r];
  }
  __syncthreads();

  if (tid < 64) {                       // 4 gates x 16 rows: cross-wave reduce + bias
    int g = tid >> 4, r = tid & 15;
    int row = g * H + wg * 16 + r;
    float s = bih[row] + bhh[row];
#pragma unroll
    for (int w = 0; w < WAVES; ++w) s += red_s[w][g][r];
    gate_s[g][r] = s;
  }
  __syncthreads();

  if (tid < 16) {                       // fused cell update (torch gate order i,f,g,o)
    int j = wg * 16 + tid;
    float ig = gate_s[0][tid], fg = gate_s[1][tid];
    float gg = gate_s[2][tid], og = gate_s[3][tid];
    float cn = sigf(fg) * c[j] + sigf(ig) * tanhf(gg);
    c[j] = cn;
    h_out[j] = sigf(og) * tanhf(cn);
  }
}

// ---------------- softmax head: logits, log_prob/entropy, sample, next input ----------------
__device__ __forceinline__ float rng01(unsigned s) {
  s ^= 2747636419u; s *= 2654435769u; s ^= s >> 16;
  s *= 2654435769u; s ^= s >> 16; s *= 2654435769u;
  return ((float)(s >> 8) + 0.5f) * (1.0f / 16777216.0f);
}

__global__ void sampler(const float* __restrict__ h, const float* __restrict__ wsw,
                        const float* __restrict__ wsb, const float* __restrict__ w_emb,
                        float* __restrict__ x_next, float* __restrict__ scal,
                        float* __restrict__ out, int flag, int kidx, int t, int last) {
  __shared__ float part[NOPS][32];
  __shared__ float logit_s[NOPS];
  __shared__ int emb_row_s;
  int tid = threadIdx.x;
  int o = tid >> 5, lane = tid & 31;

  float p = 0.f;
  for (int k = lane; k < H; k += 32) p += wsw[o * H + k] * h[k];
  part[o][lane] = p;
  __syncthreads();

  if (tid < NOPS) {
    float s = 0.f;
#pragma unroll
    for (int l = 0; l < 32; ++l) s += part[tid][l];
    s += wsb[tid];
    // logits = (tanh_constant/op_tanh_reduce) * tanh(raw / TEMP) = 1.0 * tanh(raw/5)
    logit_s[tid] = tanhf(s * 0.2f);
  }
  __syncthreads();

  if (tid == 0) {
    float m = -1e30f;
    for (int i = 0; i < NOPS; ++i) m = fmaxf(m, logit_s[i]);
    float z = 0.f;
    for (int i = 0; i < NOPS; ++i) z += __expf(logit_s[i] - m);
    float lse = m + __logf(z);
    float ent = 0.f; int op = 0; float best = -1e30f;
    for (int i = 0; i < NOPS; ++i) {
      float lpo = logit_s[i] - lse;
      ent -= __expf(lpo) * lpo;
      float u = rng01((unsigned)(t * 131 + i) * 0x9E3779B9u + 12345u);
      float sc = logit_s[i] + (-__logf(-__logf(u)));   // Gumbel-argmax categorical
      if (sc > best) { best = sc; op = i; }
    }
    if (flag) {
      scal[0] += logit_s[op] - lse;                    // log_prob accumulation
      scal[1] += ent;                                  // entropy accumulation
      out[kidx] = (float)op;                           // arc_seq (as float)
      for (int i = 0; i < NOPS; ++i) out[28 + kidx * 8 + i] = logit_s[i];
    }
    if (last) { out[252] = scal[0]; out[253] = scal[1]; }
    emb_row_s = flag ? op : NOPS;                      // anchor step -> start token
  }
  __syncthreads();

  int er = emb_row_s;
  for (int j = tid; j < H; j += blockDim.x) x_next[j] = w_emb[er * H + j];
}

// ---------------- host orchestration ----------------
extern "C" void kernel_launch(void* const* d_in, const int* in_sizes, int n_in,
                              void* d_out, int out_size, void* d_ws, size_t ws_size,
                              hipStream_t stream) {
  const float* w_emb = (const float*)d_in[0];
  const float* W_ih  = (const float*)d_in[1];
  const float* W_hh  = (const float*)d_in[2];
  const float* b_ih  = (const float*)d_in[3];
  const float* b_hh  = (const float*)d_in[4];
  const float* wsw   = (const float*)d_in[5];
  const float* wsb   = (const float*)d_in[6];
  float* out = (float*)d_out;

  const size_t WN = 2 * LAYERW;                 // elems per weight tensor (both layers)
  const size_t bf16_bytes = 2 * WN * sizeof(unsigned short);
  const size_t state_bytes = (size_t)(H + 4 * H + 2 * H + 2) * sizeof(float);
  const bool use_bf16 = (ws_size >= bf16_bytes + state_bytes + 256);

  char* ws = (char*)d_ws;
  __bf16* Wih_bf = nullptr; __bf16* Whh_bf = nullptr;
  if (use_bf16) {
    Wih_bf = (__bf16*)ws; ws += WN * sizeof(unsigned short);
    Whh_bf = (__bf16*)ws; ws += WN * sizeof(unsigned short);
  }
  float* x    = (float*)ws; ws += H * sizeof(float);
  float* hbuf = (float*)ws; ws += 4 * H * sizeof(float);  // h0 ping/pong, h1 ping/pong
  float* cbuf = (float*)ws; ws += 2 * H * sizeof(float);
  float* scal = (float*)ws;

  if (use_bf16) {
    cvt_bf16<<<4096, 256, 0, stream>>>(W_ih, Wih_bf, WN);
    cvt_bf16<<<4096, 256, 0, stream>>>(W_hh, Whh_bf, WN);
  }
  init_state<<<8, 256, 0, stream>>>(w_emb, x, hbuf, cbuf, scal);

  // static schedule: [0,0] then per node [1]*(node+1),[0]  => T=37, 28 sample steps
  int flags[64]; int T = 0;
  flags[T++] = 0; flags[T++] = 0;
  for (int node = 0; node < NNODES; ++node) {
    for (int i = 0; i <= node; ++i) flags[T++] = 1;
    flags[T++] = 0;
  }

  int kidx = 0;
  for (int t = 0; t < T; ++t) {
    float* h0_in  = hbuf + ((t    ) & 1) * H;
    float* h0_out = hbuf + ((t + 1) & 1) * H;
    float* h1_in  = hbuf + 2 * H + ((t    ) & 1) * H;
    float* h1_out = hbuf + 2 * H + ((t + 1) & 1) * H;

    if (use_bf16) {
      lstm_layer<__bf16><<<128, 512, 0, stream>>>(
          Wih_bf, Whh_bf, b_ih, b_hh, x, h0_in, cbuf, h0_out);
      lstm_layer<__bf16><<<128, 512, 0, stream>>>(
          Wih_bf + LAYERW, Whh_bf + LAYERW, b_ih + FOURH, b_hh + FOURH,
          h0_out, h1_in, cbuf + H, h1_out);
    } else {
      lstm_layer<float><<<128, 512, 0, stream>>>(
          W_ih, W_hh, b_ih, b_hh, x, h0_in, cbuf, h0_out);
      lstm_layer<float><<<128, 512, 0, stream>>>(
          W_ih + LAYERW, W_hh + LAYERW, b_ih + FOURH, b_hh + FOURH,
          h0_out, h1_in, cbuf + H, h1_out);
    }

    sampler<<<1, 256, 0, stream>>>(h1_out, wsw, wsb, w_emb, x, scal, out,
                                   flags[t], kidx, t, (t == T - 1) ? 1 : 0);
    if (flags[t]) ++kidx;
  }
}
